// QRoboticTransformer_34471407517861
// MI455X (gfx1250) — matrix operand order repaired and checked by reading, not compiled, not run on hardware
//
#include <hip/hip_runtime.h>

#define TOK    49
#define DMODEL 512
#define DHH    32
#define NHEAD  16
#define NMEM   4
#define NTOT   53      // NMEM + TOK
#define XSTR   520     // xn / q / attn_out row stride (halves)
#define KSTR   40      // k row stride (halves)
#define VSTR   72      // vT row stride (halves)

typedef __attribute__((ext_vector_type(16))) __bf16 v16bf;
typedef __attribute__((ext_vector_type(8)))  float  v8f;

union FragB { v16bf v; uint4 q[2]; unsigned short h[16]; unsigned int u[8]; };
union FragC { v8f v; float f[8]; };

// Native bf16 convert (RNE) — lowers to v_cvt_bf16_f32 / v_cvt_pk_bf16_f32.
__device__ inline unsigned short f2bf(float f) {
  __bf16 b = (__bf16)f;
  return __builtin_bit_cast(unsigned short, b);
}
__device__ inline unsigned int pack2(float a, float b) {
  return (unsigned int)f2bf(a) | ((unsigned int)f2bf(b) << 16);
}

// ---------------------------------------------------------------------------
// Pack an fp32 row-major (K x N) weight matrix into bf16 WMMA B-fragments:
// layout [nt][kt][lane][16 halves]; lane<16 holds k = kt*32+0..15 (n = nt*16+lane),
// lane>=16 holds k = kt*32+16..31. ktiles must be 16 (K=512) for load-side stride.
// ---------------------------------------------------------------------------
__global__ void pack_w_kernel(const float* __restrict__ w, unsigned short* __restrict__ out,
                              int ncols, int ntiles, int ktiles) {
  int idx = blockIdx.x * blockDim.x + threadIdx.x;
  int total = ntiles * ktiles * 32 * 16;
  if (idx >= total) return;
  int p    = idx & 15;
  int lane = (idx >> 4) & 31;
  int f    = idx >> 9;
  int kt   = f % ktiles;
  int nt   = f / ktiles;
  int kl   = p + 16 * (lane >> 4);
  int n    = nt * 16 + (lane & 15);
  int k    = kt * 32 + kl;
  out[idx] = f2bf(w[(size_t)k * ncols + n]);
}

// biasP[h][i(64)][j(64)]: j<4 (mem) -> 0; 4<=j<53 -> rel_bias[rel_idx[i][j-4]][h];
// j>=53 -> -1e30 (mask). i>=49 -> 0 (rows discarded, keep finite).
__global__ void prep_bias_kernel(const float* __restrict__ rel_bias,
                                 const int* __restrict__ rel_idx,
                                 float* __restrict__ biasP) {
  int idx = blockIdx.x * blockDim.x + threadIdx.x;
  if (idx >= NHEAD * 64 * 64) return;
  int j = idx & 63;
  int i = (idx >> 6) & 63;
  int h = idx >> 12;
  float v;
  if (j >= NTOT)                 v = -1e30f;
  else if (j < NMEM || i >= TOK) v = 0.f;
  else                           v = rel_bias[rel_idx[i * TOK + (j - NMEM)] * NHEAD + h];
  biasP[idx] = v;
}

// ---------------------------------------------------------------------------
// One workgroup (256 threads = 8 waves) per window.
// ---------------------------------------------------------------------------
__global__ __launch_bounds__(256)
void maxvit_attn_kernel(const float* __restrict__ x,
                        const float* __restrict__ gamma,
                        const float* __restrict__ beta,
                        const float* __restrict__ memkv,
                        const unsigned short* __restrict__ wqkv_p,
                        const unsigned short* __restrict__ wout_p,
                        const float* __restrict__ biasP,
                        float* __restrict__ out) {
  extern __shared__ unsigned short smem[];
  unsigned short* xns = smem;                        // 64 x 520 bf16 (xn, later attn_out)
  unsigned short* qs  = xns + 64 * XSTR;             // 64 x 520 bf16 (q, row-major, scaled)
  unsigned short* ks  = qs  + 64 * XSTR;             // [16 h][64 j][40] bf16
  unsigned short* vTs = ks  + NHEAD * 64 * KSTR;     // [16 h][32 d][72] bf16 (transposed V)

  const int tid  = threadIdx.x;
  const int lane = tid & 31;
  const int wid  = __builtin_amdgcn_readfirstlane(tid >> 5);   // wave-uniform -> SGPR
  const int hi   = lane >> 4;
  const int l15  = lane & 15;
  const int win  = blockIdx.x;
  const float scale = 0.17677669529663687f;          // 1/sqrt(32)

  // ---- Phase 0: zero pads, fill mem-kv tokens, LayerNorm -> xns (bf16) ----
  {
    unsigned int* xz = (unsigned int*)(xns + TOK * XSTR);        // rows 49..63
    for (int i = tid; i < 15 * (XSTR / 2); i += 256) xz[i] = 0u;
    for (int i = tid; i < NHEAD * 11 * (KSTR / 2); i += 256) {   // k rows j=53..63
      int h = i / (11 * (KSTR / 2));
      int r = i % (11 * (KSTR / 2));
      int j = 53 + r / (KSTR / 2);
      int c = r % (KSTR / 2);
      ((unsigned int*)ks)[(h * 64 + j) * (KSTR / 2) + c] = 0u;
    }
    for (int i = tid; i < NHEAD * DHH * 11; i += 256) {          // vT cols j=53..63
      int row = i / 11;
      int j   = 53 + i % 11;
      vTs[row * VSTR + j] = 0;
    }
    for (int i = tid; i < NHEAD * NMEM * DHH; i += 256) {        // mem tokens j=0..3
      int d = i & 31; int j = (i >> 5) & 3; int h = i >> 7;
      ks [(h * 64 + j) * KSTR + d] = f2bf(memkv[i]);
      vTs[(h * DHH + d) * VSTR + j] = f2bf(memkv[NHEAD * NMEM * DHH + i]);
    }
  }
  {
    const float* xb = x + (size_t)win * TOK * DMODEL;
    for (int t = wid; t < TOK; t += 8) {
      float4 vals[4];
      const float* xr = xb + t * DMODEL;
      float s = 0.f;
      #pragma unroll
      for (int i = 0; i < 4; ++i) {
        vals[i] = ((const float4*)xr)[i * 32 + lane];
        s += vals[i].x + vals[i].y + vals[i].z + vals[i].w;
      }
      #pragma unroll
      for (int off = 16; off >= 1; off >>= 1) s += __shfl_xor(s, off, 32);
      float mu = s * (1.f / 512.f);
      float vv = 0.f;
      #pragma unroll
      for (int i = 0; i < 4; ++i) {
        float a = vals[i].x - mu, b2 = vals[i].y - mu, c = vals[i].z - mu, d = vals[i].w - mu;
        vv += a * a + b2 * b2 + c * c + d * d;
      }
      #pragma unroll
      for (int off = 16; off >= 1; off >>= 1) vv += __shfl_xor(vv, off, 32);
      float inv = rsqrtf(vv * (1.f / 512.f) + 1e-5f);
      #pragma unroll
      for (int i = 0; i < 4; ++i) {
        int c0 = (i * 32 + lane) * 4;
        float4 g = ((const float4*)gamma)[i * 32 + lane];
        float4 b = ((const float4*)beta)[i * 32 + lane];
        uint2 pk;
        pk.x = pack2((vals[i].x - mu) * inv * g.x + b.x, (vals[i].y - mu) * inv * g.y + b.y);
        pk.y = pack2((vals[i].z - mu) * inv * g.z + b.z, (vals[i].w - mu) * inv * g.w + b.w);
        *(uint2*)(xns + t * XSTR + c0) = pk;
      }
    }
  }
  __syncthreads();

  // ---- Phase 1: QKV GEMM (64x512 @ 512x1536) via bf16 WMMA, route to q/k/vT ----
  for (int tile = wid; tile < 4 * 96; tile += 8) {
    const int mt = __builtin_amdgcn_readfirstlane(tile / 96);
    const int nt = __builtin_amdgcn_readfirstlane(tile % 96);
    FragC acc;
    #pragma unroll
    for (int r = 0; r < 8; ++r) acc.f[r] = 0.f;
    #pragma unroll 4
    for (int kt = 0; kt < 16; ++kt) {
      FragB a, b;
      const unsigned short* ar = xns + (mt * 16 + l15) * XSTR + kt * 32 + 16 * hi;
      a.q[0] = *(const uint4*)ar;
      a.q[1] = *(const uint4*)(ar + 8);
      const uint4* br = (const uint4*)wqkv_p + ((size_t)(nt * 16 + kt) * 32 + lane) * 2;
      b.q[0] = br[0];
      b.q[1] = br[1];
      acc.v = __builtin_amdgcn_wmma_f32_16x16x32_bf16(false, a.v, false, b.v,
                                                      (short)0, acc.v, false, false);
    }
    if (nt < 32) {                                   // Q: row-major, pre-scaled
      #pragma unroll
      for (int r = 0; r < 8; ++r) {
        int m = mt * 16 + r + 8 * hi;
        qs[m * XSTR + nt * 16 + l15] = f2bf(acc.f[r] * scale);
      }
    } else if (nt < 64) {                            // K: k[h][j=m+4][d] row-major
      int fk = (nt - 32) * 16 + l15;
      int h = fk >> 5, d = fk & 31;
      unsigned short* kb = ks + (h * 64 + NMEM) * KSTR + d;
      if (mt < 3) {
        #pragma unroll
        for (int r = 0; r < 8; ++r)
          kb[(mt * 16 + r + 8 * hi) * KSTR] = f2bf(acc.f[r]);
      } else if (hi == 0) {
        kb[48 * KSTR] = f2bf(acc.f[0]);              // token 48 -> j = 52
      }
    } else {                                         // V: transposed vT[h][d][j=m+4]
      int fv = (nt - 64) * 16 + l15;
      int h = fv >> 5, d = fv & 31;
      unsigned short* vr = vTs + (h * DHH + d) * VSTR + mt * 16 + 8 * hi + NMEM;
      if (mt < 3) {
        uint2 p0, p1;
        p0.x = pack2(acc.f[0], acc.f[1]); p0.y = pack2(acc.f[2], acc.f[3]);
        p1.x = pack2(acc.f[4], acc.f[5]); p1.y = pack2(acc.f[6], acc.f[7]);
        *(uint2*)vr = p0;
        *(uint2*)(vr + 4) = p1;
      } else if (hi == 0) {
        vr[0] = f2bf(acc.f[0]);                      // token 48 -> j = 52
      }
    }
  }
  __syncthreads();

  // ---- Phase 2: attention. Each wave owns 2 heads. simT = k @ qT (j x i). ----
  for (int hh = 0; hh < 2; ++hh) {
    const int h = __builtin_amdgcn_readfirstlane(wid * 2 + hh);
    for (int it = 0; it < 4; ++it) {
      FragB qf;                                      // B-frag: lane=i, k=d contiguous 16
      const unsigned short* qr = qs + (it * 16 + l15) * XSTR + h * DHH + 16 * hi;
      qf.q[0] = *(const uint4*)qr;
      qf.q[1] = *(const uint4*)(qr + 8);
      FragC s[4];
      #pragma unroll
      for (int jt = 0; jt < 4; ++jt) {
        FragB kf;                                    // A-frag: lane=row j, d contiguous 16
        const unsigned short* kr = ks + (h * 64 + jt * 16 + l15) * KSTR + 16 * hi;
        kf.q[0] = *(const uint4*)kr;
        kf.q[1] = *(const uint4*)(kr + 8);
        v8f z = {0.f, 0.f, 0.f, 0.f, 0.f, 0.f, 0.f, 0.f};
        s[jt].v = __builtin_amdgcn_wmma_f32_16x16x32_bf16(false, kf.v, false, qf.v,
                                                          (short)0, z, false, false);
      }
      // bias add + softmax over j (column = lane pair {lane, lane^16})
      int i = it * 16 + l15;
      const float4* bp = (const float4*)(biasP + ((size_t)h * 64 + i) * 64);
      float mx = -3.0e38f;
      #pragma unroll
      for (int jt = 0; jt < 4; ++jt) {
        float4 b0 = bp[jt * 4 + 2 * hi];
        float4 b1 = bp[jt * 4 + 2 * hi + 1];
        s[jt].f[0] += b0.x; s[jt].f[1] += b0.y; s[jt].f[2] += b0.z; s[jt].f[3] += b0.w;
        s[jt].f[4] += b1.x; s[jt].f[5] += b1.y; s[jt].f[6] += b1.z; s[jt].f[7] += b1.w;
        #pragma unroll
        for (int r = 0; r < 8; ++r) mx = fmaxf(mx, s[jt].f[r]);
      }
      mx = fmaxf(mx, __shfl_xor(mx, 16, 32));
      float sum = 0.f;
      #pragma unroll
      for (int jt = 0; jt < 4; ++jt) {
        #pragma unroll
        for (int r = 0; r < 8; ++r) {
          float e = __expf(s[jt].f[r] - mx);
          s[jt].f[r] = e;
          sum += e;
        }
      }
      sum += __shfl_xor(sum, 16, 32);
      float rinv = 1.f / sum;
      #pragma unroll
      for (int jt = 0; jt < 4; ++jt)
        #pragma unroll
        for (int r = 0; r < 8; ++r) s[jt].f[r] *= rinv;

      // Repack probs (accumulator layout) into B-frags (k=j): hi-split 8 -> 16
      FragB pf[2];
      #pragma unroll
      for (int st = 0; st < 2; ++st) {
        #pragma unroll
        for (int e = 0; e < 8; ++e) {
          float o0 = __shfl_xor(s[st * 2].f[e], 16, 32);
          float o1 = __shfl_xor(s[st * 2 + 1].f[e], 16, 32);
          pf[st].h[e]     = f2bf(hi ? o1 : s[st * 2].f[e]);
          pf[st].h[8 + e] = f2bf(hi ? s[st * 2 + 1].f[e] : o0);
        }
      }
      // outT = vT (16d x 64j) @ attnT (64j x 16i), two K-steps of 32
      #pragma unroll
      for (int dt = 0; dt < 2; ++dt) {
        FragC o;
        #pragma unroll
        for (int r = 0; r < 8; ++r) o.f[r] = 0.f;
        #pragma unroll
        for (int st = 0; st < 2; ++st) {
          FragB vf;                                  // A-frag: lane=row d, j contiguous 16
          const unsigned short* vr = vTs + (h * DHH + dt * 16 + l15) * VSTR + st * 32 + 16 * hi;
          vf.q[0] = *(const uint4*)vr;
          vf.q[1] = *(const uint4*)(vr + 8);
          o.v = __builtin_amdgcn_wmma_f32_16x16x32_bf16(false, vf.v, false, pf[st].v,
                                                        (short)0, o.v, false, false);
        }
        // store merged-head output row-major into xns (reused): row i, 8 contiguous cols
        unsigned short* orow = xns + i * XSTR + h * DHH + dt * 16 + 8 * hi;
        uint4 pk;
        pk.x = pack2(o.f[0], o.f[1]); pk.y = pack2(o.f[2], o.f[3]);
        pk.z = pack2(o.f[4], o.f[5]); pk.w = pack2(o.f[6], o.f[7]);
        *(uint4*)orow = pk;
      }
    }
  }
  __syncthreads();

  // ---- Phase 3: output projection (64x512 @ 512x512) -> global f32 ----
  for (int tile = wid; tile < 4 * 32; tile += 8) {
    const int mt = __builtin_amdgcn_readfirstlane(tile >> 5);
    const int nt = __builtin_amdgcn_readfirstlane(tile & 31);
    FragC acc;
    #pragma unroll
    for (int r = 0; r < 8; ++r) acc.f[r] = 0.f;
    #pragma unroll 4
    for (int kt = 0; kt < 16; ++kt) {
      FragB a, b;
      const unsigned short* ar = xns + (mt * 16 + l15) * XSTR + kt * 32 + 16 * hi;
      a.q[0] = *(const uint4*)ar;
      a.q[1] = *(const uint4*)(ar + 8);
      const uint4* br = (const uint4*)wout_p + ((size_t)(nt * 16 + kt) * 32 + lane) * 2;
      b.q[0] = br[0];
      b.q[1] = br[1];
      acc.v = __builtin_amdgcn_wmma_f32_16x16x32_bf16(false, a.v, false, b.v,
                                                      (short)0, acc.v, false, false);
    }
    float* ob = out + (size_t)win * TOK * DMODEL;
    int col = nt * 16 + l15;
    if (mt < 3) {
      #pragma unroll
      for (int r = 0; r < 8; ++r) {
        int g = mt * 16 + r + 8 * hi;
        ob[g * DMODEL + col] = acc.f[r];
      }
    } else if (hi == 0) {
      ob[48 * DMODEL + col] = acc.f[0];              // only token 48 in last tile
    }
  }
}

extern "C" void kernel_launch(void* const* d_in, const int* in_sizes, int n_in,
                              void* d_out, int out_size, void* d_ws, size_t ws_size,
                              hipStream_t stream) {
  const float* x        = (const float*)d_in[0];
  const float* ln_gamma = (const float*)d_in[1];
  const float* ln_beta  = (const float*)d_in[2];
  const float* w_qkv    = (const float*)d_in[3];
  const float* mem_kv   = (const float*)d_in[4];
  const float* w_out    = (const float*)d_in[5];
  const float* rel_bias = (const float*)d_in[6];
  const int*   rel_idx  = (const int*)d_in[7];

  // workspace: wqkv fragments (1.5MB bf16) | wout fragments (512KB) | biasP (256KB f32)
  unsigned short* wqkv_p = (unsigned short*)d_ws;
  unsigned short* wout_p = wqkv_p + 96 * 16 * 32 * 16;
  float*          biasP  = (float*)(wout_p + 32 * 16 * 32 * 16);

  int nwq = 96 * 16 * 32 * 16;
  pack_w_kernel<<<(nwq + 255) / 256, 256, 0, stream>>>(w_qkv, wqkv_p, 1536, 96, 16);
  int nwo = 32 * 16 * 32 * 16;
  pack_w_kernel<<<(nwo + 255) / 256, 256, 0, stream>>>(w_out, wout_p, 512, 32, 16);
  prep_bias_kernel<<<(NHEAD * 64 * 64 + 255) / 256, 256, 0, stream>>>(rel_bias, rel_idx, biasP);

  int windows = in_sizes[0] / (TOK * DMODEL);                    // 2048
  size_t lds_bytes = (size_t)(64 * XSTR * 2 + NHEAD * 64 * KSTR + NHEAD * DHH * VSTR) * 2; // 288768
  (void)hipFuncSetAttribute((const void*)maxvit_attn_kernel,
                            hipFuncAttributeMaxDynamicSharedMemorySize, (int)lds_bytes);
  maxvit_attn_kernel<<<windows, 256, lds_bytes, stream>>>(
      x, ln_gamma, ln_beta, mem_kv, wqkv_p, wout_p, biasP, (float*)d_out);
}
